// AttnBlock_79293686219024
// MI455X (gfx1250) — matrix operand order repaired and checked by reading, not compiled, not run on hardware
//
#include <hip/hip_runtime.h>
#include <hip/hip_bf16.h>

#define C_DIM 512
#define N_SP  4096
#define B_DIM 4
#define GROUPS 32
#define CPG   16
#define GN_EPS 1e-5f
#define SROW  4100   // padded fp32 stride for S rows in LDS (16B aligned, conflict-free)

typedef __attribute__((ext_vector_type(16))) __bf16 bf16x16;
typedef __attribute__((ext_vector_type(8)))  __bf16 bf16x8;
typedef __attribute__((ext_vector_type(8)))  float  f32x8;

__device__ __forceinline__ f32x8 wmma_bf16(bf16x16 a, bf16x16 b, f32x8 c) {
  // D = A(16x32 bf16) x B(32x16 bf16) + C(16x16 f32)
  return __builtin_amdgcn_wmma_f32_16x16x32_bf16(false, a, false, b, (short)0, c, false, false);
}

__device__ __forceinline__ f32x8 zero8() {
  f32x8 z = {0.f,0.f,0.f,0.f,0.f,0.f,0.f,0.f};
  return z;
}

// A-operand (16x32, row owned by lane lm): K elements klo+0..7 and 16+klo+0..7
__device__ __forceinline__ bf16x16 ld_afrag(const __bf16* row, int k0, int klo) {
  bf16x8 lo = *reinterpret_cast<const bf16x8*>(row + k0 + klo);
  bf16x8 hi = *reinterpret_cast<const bf16x8*>(row + k0 + klo + 16);
  bf16x16 r;
#pragma unroll
  for (int i = 0; i < 8; ++i) { r[i] = lo[i]; r[8 + i] = hi[i]; }
  return r;
}

// B-operand (32x16, column owned by lane lm): K elements 8*hi+0..7 and 16+8*hi+0..7
__device__ __forceinline__ bf16x16 ld_bfrag(const __bf16* colrow, int k0, int hi16) {
  const int klo = hi16 >> 1;
  bf16x8 lo = *reinterpret_cast<const bf16x8*>(colrow + k0 + klo);
  bf16x8 hi = *reinterpret_cast<const bf16x8*>(colrow + k0 + klo + 16);
  bf16x16 r;
#pragma unroll
  for (int i = 0; i < 8; ++i) { r[i] = lo[i]; r[8 + i] = hi[i]; }
  return r;
}

// ---------------- GroupNorm stats: one block per (b, group); data contiguous ----------------
__global__ __launch_bounds__(256) void gn_stats_kernel(const float* __restrict__ x,
                                                       float* __restrict__ stats) {
  const int gid = blockIdx.x;                    // b*GROUPS + g
  const float* p = x + (size_t)gid * (CPG * N_SP);
  float s = 0.f, sq = 0.f;
  for (int i = threadIdx.x; i < (CPG * N_SP) / 4; i += 256) {
    float4 v = reinterpret_cast<const float4*>(p)[i];
    s  += v.x + v.y + v.z + v.w;
    sq += v.x*v.x + v.y*v.y + v.z*v.z + v.w*v.w;
  }
  __shared__ float rs[256], rq[256];
  rs[threadIdx.x] = s; rq[threadIdx.x] = sq;
  __syncthreads();
  for (int o = 128; o > 0; o >>= 1) {
    if (threadIdx.x < o) { rs[threadIdx.x] += rs[threadIdx.x + o]; rq[threadIdx.x] += rq[threadIdx.x + o]; }
    __syncthreads();
  }
  if (threadIdx.x == 0) {
    const float inv = 1.f / (float)(CPG * N_SP);
    float mean = rs[0] * inv;
    float var  = rq[0] * inv - mean * mean;
    stats[2*gid]   = mean;
    stats[2*gid+1] = rsqrtf(var + GN_EPS);
  }
}

// ---------------- normalize + transpose [B,C,N] -> bf16 [B,N,C] via LDS tile ----------------
__global__ __launch_bounds__(256) void gn_apply_kernel(const float* __restrict__ x,
    const float* __restrict__ stats, const float* __restrict__ gamma,
    const float* __restrict__ beta, __bf16* __restrict__ h) {
  const int b = blockIdx.z, ct = blockIdx.y, nt = blockIdx.x;
  const int nbase = nt * 256, cbase = ct * 16;
  __shared__ float tile[16][257];
  const int tid = threadIdx.x;
  const int sid = b * GROUPS + (cbase >> 4);     // one 16-channel tile == one group
  const float mean = stats[2*sid], rstd = stats[2*sid+1];
#pragma unroll
  for (int cc = 0; cc < 16; ++cc) {
    const int c = cbase + cc;
    const float g = gamma[c] * rstd;
    const float o = beta[c] - mean * g;
    const float v = x[((size_t)(b * C_DIM + c)) * N_SP + nbase + tid];
    tile[cc][tid] = v * g + o;
  }
  __syncthreads();
  const int cc = tid & 15, rr = tid >> 4;
#pragma unroll
  for (int it = 0; it < 16; ++it) {
    const int n = nbase + it * 16 + rr;
    h[((size_t)b * N_SP + n) * C_DIM + cbase + cc] = (__bf16)tile[cc][it * 16 + rr];
  }
}

// ---------------- fp32 -> bf16 weight conversion ----------------
__global__ __launch_bounds__(256) void cvt_bf16_kernel(const float* __restrict__ src,
                                                       __bf16* __restrict__ dst, int n) {
  for (int i = blockIdx.x * 256 + threadIdx.x; i < n; i += gridDim.x * 256)
    dst[i] = (__bf16)src[i];
}

// ---------------- projection GEMM: out[n][c] = (A[n][:] . W[c][:]) + bias[c], x scale --------
__global__ __launch_bounds__(256) void proj_kernel(const __bf16* __restrict__ A,   // [B*N, C]
    const __bf16* __restrict__ W,    // [C, C] row-major (row = out channel)
    const float* __restrict__ bias, __bf16* __restrict__ out, float scale, int transposeOut) {
  const int tid = threadIdx.x, wave = tid >> 5, lane = tid & 31;
  const int lm = lane & 15, hi = lane >> 4, klo = hi * 8, hi16 = hi * 16;
  const int r0 = blockIdx.x * 32;
  const int cbase = wave * 64;
  f32x8 acc[2][4];
#pragma unroll
  for (int h = 0; h < 2; ++h)
#pragma unroll
    for (int t = 0; t < 4; ++t) acc[h][t] = zero8();

  const __bf16* arow0 = A + (size_t)(r0 + lm) * C_DIM;
  const __bf16* arow1 = arow0 + (size_t)16 * C_DIM;
  for (int k0 = 0; k0 < C_DIM; k0 += 32) {
    bf16x16 a0 = ld_afrag(arow0, k0, klo);
    bf16x16 a1 = ld_afrag(arow1, k0, klo);
#pragma unroll
    for (int t = 0; t < 4; ++t) {
      const __bf16* wrow = W + (size_t)(cbase + t * 16 + lm) * C_DIM;
      bf16x16 bf = ld_bfrag(wrow, k0, hi16);
      acc[0][t] = wmma_bf16(a0, bf, acc[0][t]);
      acc[1][t] = wmma_bf16(a1, bf, acc[1][t]);
    }
  }
#pragma unroll
  for (int t = 0; t < 4; ++t) {
    const int col = cbase + t * 16 + lm;
    const float bv = bias[col];
#pragma unroll
    for (int h = 0; h < 2; ++h) {
#pragma unroll
      for (int r = 0; r < 8; ++r) {
        const int rg = r0 + h * 16 + r + 8 * hi;
        const float val = (acc[h][t][r] + bv) * scale;
        if (!transposeOut) {
          out[(size_t)rg * C_DIM + col] = (__bf16)val;
        } else {
          const int bb = rg >> 12, nn = rg & (N_SP - 1);
          out[((size_t)bb * C_DIM + col) * N_SP + nn] = (__bf16)val;
        }
      }
    }
  }
}

// ---------------- attention: one block per 16-query band; full fp32 S band in LDS ----------
__global__ __launch_bounds__(256) void attn_kernel(const __bf16* __restrict__ Q,  // [B,N,C], pre-scaled
    const __bf16* __restrict__ K,   // [B,N,C]
    const __bf16* __restrict__ Vt,  // [B,C,N]
    __bf16* __restrict__ O) {       // [B,N,C]
  extern __shared__ float smem[];
  float* S      = smem;                 // [16][SROW]
  float* wm     = smem + 16 * SROW;     // [8][16] per-wave running max
  float* wl     = wm + 128;             // [8][16] per-wave running sum
  float* linv_s = wl + 128;             // [16]

  const int tid = threadIdx.x, wave = tid >> 5, lane = tid & 31;
  const int lm = lane & 15, hi = lane >> 4, klo = hi * 8, hi16 = hi * 16;
  const int b = blockIdx.x >> 8;
  const int qbase = (blockIdx.x & 255) * 16;
  const __bf16* Qb = Q + ((size_t)b * N_SP + qbase) * C_DIM;
  const __bf16* Kb = K + (size_t)b * N_SP * C_DIM;
  const __bf16* Vb = Vt + (size_t)b * C_DIM * N_SP;

  // Preload all 16 Q B-fragments (column = query lm), reused across 512 keys.
  bf16x16 qf[16];
  const __bf16* qrow = Qb + (size_t)lm * C_DIM;
#pragma unroll
  for (int c = 0; c < 16; ++c) qf[c] = ld_bfrag(qrow, c * 32, hi16);

  // Phase 1: S^T tiles (D rows = keys, cols = queries) + online softmax stats.
  float mr = -3.0e38f, lr = 0.f;
  for (int it = 0; it < 32; ++it) {
    const int kb = wave * 512 + it * 16;
    f32x8 acc = zero8();
    const __bf16* krow = Kb + (size_t)(kb + lm) * C_DIM;
#pragma unroll
    for (int c = 0; c < 16; ++c) {
      bf16x16 af = ld_afrag(krow, c * 32, klo);
      acc = wmma_bf16(af, qf[c], acc);
    }
    float* dst = S + (size_t)lm * SROW + kb + 8 * hi;   // lane owns query lm, 8 consecutive keys
    *reinterpret_cast<float4*>(dst)     = make_float4(acc[0], acc[1], acc[2], acc[3]);
    *reinterpret_cast<float4*>(dst + 4) = make_float4(acc[4], acc[5], acc[6], acc[7]);
    float tm = acc[0];
#pragma unroll
    for (int r = 1; r < 8; ++r) tm = fmaxf(tm, acc[r]);
    const float mnew = fmaxf(mr, tm);
    float lsum = 0.f;
#pragma unroll
    for (int r = 0; r < 8; ++r) lsum += __expf(acc[r] - mnew);
    lr = lr * __expf(mr - mnew) + lsum;
    mr = mnew;
  }
  {
    const float mo = __shfl_xor(mr, 16);
    const float lo = __shfl_xor(lr, 16);
    const float m2 = fmaxf(mr, mo);
    const float l2 = lr * __expf(mr - m2) + lo * __expf(mo - m2);
    if (lane < 16) { wm[wave * 16 + lane] = m2; wl[wave * 16 + lane] = l2; }
  }
  __syncthreads();

  // Final stats for query q = lm (computed redundantly by every thread).
  float mfin = -3.0e38f;
#pragma unroll
  for (int w = 0; w < 8; ++w) mfin = fmaxf(mfin, wm[w * 16 + lm]);
  float lfin = 0.f;
#pragma unroll
  for (int w = 0; w < 8; ++w) lfin += wl[w * 16 + lm] * __expf(wm[w * 16 + lm] - mfin);
  if (tid < 16) linv_s[tid] = 1.f / lfin;
  __syncthreads();

  // Phase 2: O slice = P @ V, P built on the fly from LDS S (lane owns exactly query row lm).
  const int cslice = wave * 64;
  f32x8 acc2[4];
#pragma unroll
  for (int t = 0; t < 4; ++t) acc2[t] = zero8();
  for (int kt = 0; kt < 128; ++kt) {
    const int kb2 = kt * 32;
    const float* sp = S + (size_t)lm * SROW + kb2 + klo;
    float4 s0 = *reinterpret_cast<const float4*>(sp);
    float4 s1 = *reinterpret_cast<const float4*>(sp + 4);
    float4 s2 = *reinterpret_cast<const float4*>(sp + 16);
    float4 s3 = *reinterpret_cast<const float4*>(sp + 20);
    const float sv[16] = {s0.x, s0.y, s0.z, s0.w, s1.x, s1.y, s1.z, s1.w,
                          s2.x, s2.y, s2.z, s2.w, s3.x, s3.y, s3.z, s3.w};
    bf16x16 pa;
#pragma unroll
    for (int i = 0; i < 16; ++i) pa[i] = (__bf16)__expf(sv[i] - mfin);
#pragma unroll
    for (int t = 0; t < 4; ++t) {
      const __bf16* vrow = Vb + (size_t)(cslice + t * 16 + lm) * N_SP;
      bf16x16 bf = ld_bfrag(vrow, kb2, hi16);
      acc2[t] = wmma_bf16(pa, bf, acc2[t]);
    }
  }
  float li[8];
#pragma unroll
  for (int r = 0; r < 8; ++r) li[r] = linv_s[r + 8 * hi];
#pragma unroll
  for (int t = 0; t < 4; ++t) {
    const int col = cslice + t * 16 + lm;
#pragma unroll
    for (int r = 0; r < 8; ++r) {
      const int q = qbase + r + 8 * hi;
      O[((size_t)b * N_SP + q) * C_DIM + col] = (__bf16)(acc2[t][r] * li[r]);
    }
  }
}

// ---------------- output projection + residual: out[b][c][n] = x + O@Wo^T + bo -------------
__global__ __launch_bounds__(256) void out_proj_kernel(const __bf16* __restrict__ A,  // [B*N, C]
    const __bf16* __restrict__ W, const float* __restrict__ bias,
    const float* __restrict__ x, float* __restrict__ out) {
  const int tid = threadIdx.x, wave = tid >> 5, lane = tid & 31;
  const int lm = lane & 15, hi = lane >> 4, klo = hi * 8, hi16 = hi * 16;
  const int r0 = blockIdx.x * 32;
  const int cbase = wave * 64;
  f32x8 acc[2][4];
#pragma unroll
  for (int h = 0; h < 2; ++h)
#pragma unroll
    for (int t = 0; t < 4; ++t) acc[h][t] = zero8();

  const __bf16* arow0 = A + (size_t)(r0 + lm) * C_DIM;
  const __bf16* arow1 = arow0 + (size_t)16 * C_DIM;
  for (int k0 = 0; k0 < C_DIM; k0 += 32) {
    bf16x16 a0 = ld_afrag(arow0, k0, klo);
    bf16x16 a1 = ld_afrag(arow1, k0, klo);
#pragma unroll
    for (int t = 0; t < 4; ++t) {
      const __bf16* wrow = W + (size_t)(cbase + t * 16 + lm) * C_DIM;
      bf16x16 bf = ld_bfrag(wrow, k0, hi16);
      acc[0][t] = wmma_bf16(a0, bf, acc[0][t]);
      acc[1][t] = wmma_bf16(a1, bf, acc[1][t]);
    }
  }
  const int b  = r0 >> 12;
  const int nb = r0 & (N_SP - 1);
#pragma unroll
  for (int t = 0; t < 4; ++t) {
    const int col = cbase + t * 16 + lm;
    const float bv = bias[col];
#pragma unroll
    for (int h = 0; h < 2; ++h) {
      const size_t idx = ((size_t)b * C_DIM + col) * N_SP + nb + h * 16 + 8 * hi;
      float4 xa = *reinterpret_cast<const float4*>(x + idx);
      float4 xb = *reinterpret_cast<const float4*>(x + idx + 4);
      float4 o0 = make_float4(acc[h][t][0] + bv + xa.x, acc[h][t][1] + bv + xa.y,
                              acc[h][t][2] + bv + xa.z, acc[h][t][3] + bv + xa.w);
      float4 o1 = make_float4(acc[h][t][4] + bv + xb.x, acc[h][t][5] + bv + xb.y,
                              acc[h][t][6] + bv + xb.z, acc[h][t][7] + bv + xb.w);
      *reinterpret_cast<float4*>(out + idx)     = o0;
      *reinterpret_cast<float4*>(out + idx + 4) = o1;
    }
  }
}

extern "C" void kernel_launch(void* const* d_in, const int* in_sizes, int n_in,
                              void* d_out, int out_size, void* d_ws, size_t ws_size,
                              hipStream_t stream) {
  (void)in_sizes; (void)n_in; (void)out_size; (void)ws_size;
  const float* x        = (const float*)d_in[0];
  const float* gn_scale = (const float*)d_in[1];
  const float* gn_bias  = (const float*)d_in[2];
  const float* wq = (const float*)d_in[3]; const float* bq = (const float*)d_in[4];
  const float* wk = (const float*)d_in[5]; const float* bk = (const float*)d_in[6];
  const float* wv = (const float*)d_in[7]; const float* bv = (const float*)d_in[8];
  const float* wo = (const float*)d_in[9]; const float* bo = (const float*)d_in[10];
  float* out = (float*)d_out;

  char* ws = (char*)d_ws;
  size_t off = 0;
  auto alloc = [&](size_t bytes) -> void* {
    off = (off + 255) & ~(size_t)255;
    void* p = ws + off;
    off += bytes;
    return p;
  };
  const size_t bnc = (size_t)B_DIM * N_SP * C_DIM;
  float*  stats = (float*) alloc(128 * 2 * sizeof(float));
  __bf16* hbf   = (__bf16*)alloc(bnc * 2);
  __bf16* qbf   = (__bf16*)alloc(bnc * 2);
  __bf16* kbf   = (__bf16*)alloc(bnc * 2);
  __bf16* vbfT  = (__bf16*)alloc(bnc * 2);
  __bf16* obf   = (__bf16*)alloc(bnc * 2);
  __bf16* wqb   = (__bf16*)alloc((size_t)C_DIM * C_DIM * 2);
  __bf16* wkb   = (__bf16*)alloc((size_t)C_DIM * C_DIM * 2);
  __bf16* wvb   = (__bf16*)alloc((size_t)C_DIM * C_DIM * 2);
  __bf16* wob   = (__bf16*)alloc((size_t)C_DIM * C_DIM * 2);

  gn_stats_kernel<<<B_DIM * GROUPS, 256, 0, stream>>>(x, stats);
  gn_apply_kernel<<<dim3(N_SP / 256, C_DIM / 16, B_DIM), 256, 0, stream>>>(x, stats, gn_scale, gn_bias, hbf);

  cvt_bf16_kernel<<<256, 256, 0, stream>>>(wq, wqb, C_DIM * C_DIM);
  cvt_bf16_kernel<<<256, 256, 0, stream>>>(wk, wkb, C_DIM * C_DIM);
  cvt_bf16_kernel<<<256, 256, 0, stream>>>(wv, wvb, C_DIM * C_DIM);
  cvt_bf16_kernel<<<256, 256, 0, stream>>>(wo, wob, C_DIM * C_DIM);

  // softmax scale C^-0.5 baked into Q
  proj_kernel<<<(B_DIM * N_SP) / 32, 256, 0, stream>>>(hbf, wqb, bq, qbf, 0.04419417382415922f, 0);
  proj_kernel<<<(B_DIM * N_SP) / 32, 256, 0, stream>>>(hbf, wkb, bk, kbf, 1.0f, 0);
  proj_kernel<<<(B_DIM * N_SP) / 32, 256, 0, stream>>>(hbf, wvb, bv, vbfT, 1.0f, 1);  // V transposed

  const size_t smem = (size_t)(16 * SROW + 128 + 128 + 16) * sizeof(float);  // ~263 KB < 320 KB WGP LDS
  attn_kernel<<<B_DIM * (N_SP / 16), 256, smem, stream>>>(qbf, kbf, vbfT, obf);

  out_proj_kernel<<<(B_DIM * N_SP) / 32, 256, 0, stream>>>(obf, wob, bo, x, out);
}
